// D_LIEG_37039797960739
// MI455X (gfx1250) — matrix operand order repaired and checked by reading, not compiled, not run on hardware
//
#include <hip/hip_runtime.h>
#include <cstddef>

// ---------------- problem constants ----------------
#define SP    1024
#define HHW   16384      // 128*128
#define IMG   128
#define BAND  224
#define DIM   256
#define HEADS 8
#define DH    64
#define HID   512
#define NS    6
#define ATT_SCALE 0.125f // 64^-0.5

// ---------------- WMMA plumbing (CDNA5 gfx1250, wave32) ----------------
typedef __attribute__((ext_vector_type(16))) __bf16 v16bf;
typedef __attribute__((ext_vector_type(8)))  float  v8f;

#define LDSW 40   // LDS row stride in 16-bit units: 32 K + 8 pad, keeps 16B alignment

// pack two f32 -> packed bf16x2; prefer the single-instruction packed convert
__device__ __forceinline__ unsigned pk2bf(float a, float b) {
#if __has_builtin(__builtin_amdgcn_cvt_pk_bf16_f32)
  auto r = __builtin_amdgcn_cvt_pk_bf16_f32(a, b);
  unsigned u;
  __builtin_memcpy(&u, &r, 4);
  return u;
#else
  union { __bf16 h[2]; unsigned u; } cv;
  cv.h[0] = (__bf16)a;
  cv.h[1] = (__bf16)b;
  return cv.u;
#endif
}

// A fragment, 16x32 bf16 (M x K). Per ISA: lanes 0-15 / 16-31 => K-halves of 8.
__device__ __forceinline__ v16bf load_frag_a(const unsigned short* As, int mw, int lane) {
  int half = lane >> 4;
  int m = (lane & 15) + mw;
  union { uint4 q[2]; v16bf v; } u;
  u.q[0] = *(const uint4*)(As + m * LDSW + half * 8);
  u.q[1] = *(const uint4*)(As + m * LDSW + 16 + half * 8);
  return u.v;
}

// B fragment, 32x16 bf16 (K x N), LDS holds B column-major: Bs[n][k].
__device__ __forceinline__ v16bf load_frag_b(const unsigned short* Bs, int nw, int lane) {
  int half = lane >> 4;
  int n = (lane & 15) + nw;
  union { uint4 q[2]; v16bf v; } u;
  u.q[0] = *(const uint4*)(Bs + n * LDSW + half * 8);
  u.q[1] = *(const uint4*)(Bs + n * LDSW + 16 + half * 8);
  return u.v;
}

__device__ __forceinline__ v8f wmma_bf16(v16bf a, v16bf b, v8f c) {
  return __builtin_amdgcn_wmma_f32_16x16x32_bf16(false, a, false, b, (short)0, c, false, false);
}

__device__ __forceinline__ float gelu_exact(float x) {
  return 0.5f * x * (1.0f + erff(x * 0.70710678118654752f));
}

// ---------------- generic batched GEMM: C = act(alpha*A@B + bias + addm) ----------------
// Block tile 64(M) x 128(N), 8 waves in 2x4, each wave 32x32 (4 WMMAs / k-step).
// A: MxK row-major (lda, batch stride sA)
// B: KxN row-major (transB=0) or NxK row-major (transB=1), (ldb, sB)
// bias: per-N or null; addm: MxN (ldadd, sAdd) or null
// act: 0 none, 1 relu, 2 gelu(exact)
__global__ __launch_bounds__(256) void gemm_bf16_kernel(
    const float* __restrict__ A, int lda, long long sA,
    const float* __restrict__ B, int ldb, long long sB, int transB,
    float* __restrict__ C, int ldc, long long sC,
    const float* __restrict__ bias,
    const float* __restrict__ addm, int ldadd, long long sAdd,
    float alpha, int act, int M, int N, int K) {
  __shared__ __align__(16) unsigned short As[64 * LDSW];
  __shared__ __align__(16) unsigned short Bs[128 * LDSW];

  const int t = threadIdx.x;
  const int lane = t & 31;
  const int wid = t >> 5;
  const int m0w = (wid & 1) * 32;   // wave's M offset in tile
  const int n0w = (wid >> 1) * 32;  // wave's N offset in tile
  const int m0 = blockIdx.y * 64;
  const int n0 = blockIdx.x * 128;
  const int bz = blockIdx.z;

  const float* Ab = A + (size_t)bz * sA;
  const float* Bb = B + (size_t)bz * sB;

  v8f acc00 = {}, acc01 = {}, acc10 = {}, acc11 = {};

  // staging thread mappings
  const int ai = t >> 2, akk = (t & 3) << 3;   // A: 64 rows x 32 k, 8 elems/thread
  const int bn = t >> 1, bkk = (t & 1) << 4;   // B: 128 n x 32 k, 16 elems/thread

  for (int k0 = 0; k0 < K; k0 += 32) {
    __syncthreads();
    {  // stage A tile 64x32 -> one ds_store_b128 per thread
      int m = m0 + ai;
      uint4 v = {0u, 0u, 0u, 0u};
      if (m < M) {
        const float* p = Ab + (size_t)m * lda + k0 + akk;
        float4 f0 = *(const float4*)p;
        float4 f1 = *(const float4*)(p + 4);
        v.x = pk2bf(f0.x, f0.y); v.y = pk2bf(f0.z, f0.w);
        v.z = pk2bf(f1.x, f1.y); v.w = pk2bf(f1.z, f1.w);
        if (k0 + 32 < K) __builtin_prefetch(p + 32, 0, 1);  // speculative next K-slice
      }
      *(uint4*)(As + ai * LDSW + akk) = v;
    }
    {  // stage B tile 32x128 (stored Bs[n][k]) -> two ds_store_b128 per thread
      int n = n0 + bn;
      uint4 v0 = {0u, 0u, 0u, 0u}, v1 = {0u, 0u, 0u, 0u};
      if (n < N) {
        if (transB) {
          const float* p = Bb + (size_t)n * ldb + k0 + bkk;
          float4 f0 = *(const float4*)p;
          float4 f1 = *(const float4*)(p + 4);
          float4 f2 = *(const float4*)(p + 8);
          float4 f3 = *(const float4*)(p + 12);
          v0.x = pk2bf(f0.x, f0.y); v0.y = pk2bf(f0.z, f0.w);
          v0.z = pk2bf(f1.x, f1.y); v0.w = pk2bf(f1.z, f1.w);
          v1.x = pk2bf(f2.x, f2.y); v1.y = pk2bf(f2.z, f2.w);
          v1.z = pk2bf(f3.x, f3.y); v1.w = pk2bf(f3.z, f3.w);
        } else {
          float f[16];
          #pragma unroll
          for (int j = 0; j < 16; ++j)
            f[j] = Bb[(size_t)(k0 + bkk + j) * ldb + n];
          v0.x = pk2bf(f[0], f[1]);   v0.y = pk2bf(f[2], f[3]);
          v0.z = pk2bf(f[4], f[5]);   v0.w = pk2bf(f[6], f[7]);
          v1.x = pk2bf(f[8], f[9]);   v1.y = pk2bf(f[10], f[11]);
          v1.z = pk2bf(f[12], f[13]); v1.w = pk2bf(f[14], f[15]);
        }
      }
      *(uint4*)(Bs + bn * LDSW + bkk) = v0;
      *(uint4*)(Bs + bn * LDSW + bkk + 8) = v1;
    }
    __syncthreads();
    v16bf a0 = load_frag_a(As, m0w, lane);
    v16bf a1 = load_frag_a(As, m0w + 16, lane);
    v16bf b0 = load_frag_b(Bs, n0w, lane);
    v16bf b1 = load_frag_b(Bs, n0w + 16, lane);
    acc00 = wmma_bf16(a0, b0, acc00);
    acc01 = wmma_bf16(a0, b1, acc01);
    acc10 = wmma_bf16(a1, b0, acc10);
    acc11 = wmma_bf16(a1, b1, acc11);
  }

  // epilogue (C/D layout: lane n=lane&15, M=8*(lane>>4)+r)
  const int half = lane >> 4, nn = lane & 15;
  #pragma unroll
  for (int mi = 0; mi < 2; ++mi) {
    #pragma unroll
    for (int ni = 0; ni < 2; ++ni) {
      v8f acc = mi ? (ni ? acc11 : acc10) : (ni ? acc01 : acc00);
      int n = n0 + n0w + ni * 16 + nn;
      #pragma unroll
      for (int r = 0; r < 8; ++r) {
        int m = m0 + m0w + mi * 16 + half * 8 + r;
        if (m < M && n < N) {
          float val = alpha * acc[r];
          if (bias) val += bias[n];
          if (addm) val += addm[(size_t)bz * sAdd + (size_t)m * ldadd + n];
          if (act == 1) val = fmaxf(val, 0.0f);
          else if (act == 2) val = gelu_exact(val);
          C[(size_t)bz * sC + (size_t)m * ldc + n] = val;
        }
      }
    }
  }
}

// ---------------- conv 3x3 SAME, NCHW (N=1), implicit GEMM ----------------
// X: Cin x (128*128), Wt: (Cout,Cin,3,3), Y: Cout x (128*128)
// Same 64(pixels) x 128(Cout) tiling as the GEMM.
__global__ __launch_bounds__(256) void conv3x3_bf16_kernel(
    const float* __restrict__ X, const float* __restrict__ Wt,
    const float* __restrict__ bias, float* __restrict__ Y, int Cin, int Cout) {
  __shared__ __align__(16) unsigned short As[64 * LDSW];
  __shared__ __align__(16) unsigned short Bs[128 * LDSW];

  const int t = threadIdx.x;
  const int lane = t & 31;
  const int wid = t >> 5;
  const int m0w = (wid & 1) * 32;
  const int n0w = (wid >> 1) * 32;
  const int m0 = blockIdx.y * 64;   // pixel tile
  const int n0 = blockIdx.x * 128;  // out-channel tile

  v8f acc00 = {}, acc01 = {}, acc10 = {}, acc11 = {};

  const int ai = t >> 2, akk = (t & 3) << 3;
  const int bn = t >> 1, bkk = (t & 1) << 4;

  for (int tap = 0; tap < 9; ++tap) {
    const int dy = tap / 3 - 1, dx = tap % 3 - 1;
    for (int kc = 0; kc < Cin; kc += 32) {
      __syncthreads();
      {  // A: shifted input pixels (M=pixels, K=channels), strided per-channel gather
        int p = m0 + ai;
        int py = p >> 7, px = p & 127;
        int sy = py + dy, sx = px + dx;
        bool ok = (sy >= 0) && (sy < IMG) && (sx >= 0) && (sx < IMG);
        int spix = (sy << 7) + sx;
        uint4 v = {0u, 0u, 0u, 0u};
        if (ok) {
          const float* p0 = X + (size_t)(kc + akk) * HHW + spix;
          float f[8];
          #pragma unroll
          for (int j = 0; j < 8; ++j) f[j] = p0[(size_t)j * HHW];
          v.x = pk2bf(f[0], f[1]); v.y = pk2bf(f[2], f[3]);
          v.z = pk2bf(f[4], f[5]); v.w = pk2bf(f[6], f[7]);
        }
        *(uint4*)(As + ai * LDSW + akk) = v;
      }
      {  // B: weights W[co][ci][tap]
        int co = n0 + bn;
        uint4 v0 = {0u, 0u, 0u, 0u}, v1 = {0u, 0u, 0u, 0u};
        if (co < Cout) {
          const float* p0 = Wt + ((size_t)co * Cin + kc + bkk) * 9 + tap;
          float f[16];
          #pragma unroll
          for (int j = 0; j < 16; ++j) f[j] = p0[j * 9];
          v0.x = pk2bf(f[0], f[1]);   v0.y = pk2bf(f[2], f[3]);
          v0.z = pk2bf(f[4], f[5]);   v0.w = pk2bf(f[6], f[7]);
          v1.x = pk2bf(f[8], f[9]);   v1.y = pk2bf(f[10], f[11]);
          v1.z = pk2bf(f[12], f[13]); v1.w = pk2bf(f[14], f[15]);
        }
        *(uint4*)(Bs + bn * LDSW + bkk) = v0;
        *(uint4*)(Bs + bn * LDSW + bkk + 8) = v1;
      }
      __syncthreads();
      v16bf a0 = load_frag_a(As, m0w, lane);
      v16bf a1 = load_frag_a(As, m0w + 16, lane);
      v16bf b0 = load_frag_b(Bs, n0w, lane);
      v16bf b1 = load_frag_b(Bs, n0w + 16, lane);
      acc00 = wmma_bf16(a0, b0, acc00);
      acc01 = wmma_bf16(a0, b1, acc01);
      acc10 = wmma_bf16(a1, b0, acc10);
      acc11 = wmma_bf16(a1, b1, acc11);
    }
  }

  const int half = lane >> 4, nn = lane & 15;
  #pragma unroll
  for (int mi = 0; mi < 2; ++mi) {
    #pragma unroll
    for (int ni = 0; ni < 2; ++ni) {
      v8f acc = mi ? (ni ? acc11 : acc10) : (ni ? acc01 : acc00);
      int co = n0 + n0w + ni * 16 + nn;
      #pragma unroll
      for (int r = 0; r < 8; ++r) {
        int p = m0 + m0w + mi * 16 + half * 8 + r;
        if (co < Cout)
          Y[(size_t)co * HHW + p] = acc[r] + bias[co];
      }
    }
  }
}

// ---------------- support kernels ----------------
__global__ void zero_f_kernel(float* p, int n) {
  int i = blockIdx.x * blockDim.x + threadIdx.x;
  if (i < n) p[i] = 0.0f;
}
__global__ void zero_i_kernel(int* p, int n) {
  int i = blockIdx.x * blockDim.x + threadIdx.x;
  if (i < n) p[i] = 0;
}
// pixel -> superpixel from one-hot Q, plus member counts
__global__ void q_index_kernel(const float* __restrict__ Q, int* __restrict__ spidx,
                               int* __restrict__ counts) {
  int p = blockIdx.x * blockDim.x + threadIdx.x;
  if (p >= HHW) return;
  const float* row = Q + (size_t)p * SP;
  int sp = 0;
  for (int c = 0; c < SP; ++c)
    if (row[c] > 0.5f) { sp = c; break; }
  spidx[p] = sp;
  atomicAdd(&counts[sp], 1);
}
// acc[sp][c] += X[c][p]   (X channel-major C x HW)
__global__ void pool_accum_kernel(const float* __restrict__ X, const int* __restrict__ spidx,
                                  float* __restrict__ acc, int C) {
  int idx = blockIdx.x * blockDim.x + threadIdx.x;
  if (idx >= C * HHW) return;
  int c = idx >> 14, p = idx & (HHW - 1);
  atomicAdd(&acc[(size_t)spidx[p] * C + c], X[(size_t)c * HHW + p]);
}
__global__ void ct_div_kernel(const float* __restrict__ acc, const int* __restrict__ counts,
                              float* __restrict__ o, int C) {
  int idx = blockIdx.x * blockDim.x + threadIdx.x;
  if (idx >= SP * C) return;
  o[idx] = acc[idx] / (float)counts[idx / C];
}
// cat[sp] = relu(concat(p0[sp], p1[sp]))
__global__ void build_cat_kernel(float* __restrict__ cat, const float* __restrict__ p0, int C0,
                                 const float* __restrict__ p1, int C1) {
  int W = C0 + C1;
  int idx = blockIdx.x * blockDim.x + threadIdx.x;
  if (idx >= SP * W) return;
  int sp = idx / W, j = idx - sp * W;
  float v = (j < C0) ? p0[(size_t)sp * C0 + j] : p1[(size_t)sp * C1 + (j - C0)];
  cat[idx] = fmaxf(v, 0.0f);
}
__global__ void addpos_kernel(float* __restrict__ x, const float* __restrict__ p, int n) {
  int i = blockIdx.x * blockDim.x + threadIdx.x;
  if (i < n) x[i] += p[i];
}
__global__ void absdiff_kernel(const float* __restrict__ a, const float* __restrict__ b,
                               float* __restrict__ d, int n) {
  int i = blockIdx.x * blockDim.x + threadIdx.x;
  if (i < n) d[i] = fabsf(a[i] - b[i]);
}
// LayerNorm over DIM=256, one wave32 per row, 8 rows per 256-thread block
__global__ void ln_kernel(const float* __restrict__ x, const float* __restrict__ g,
                          const float* __restrict__ b, float* __restrict__ o) {
  int row = blockIdx.x * 8 + (threadIdx.x >> 5);
  int lane = threadIdx.x & 31;
  const float* xr = x + (size_t)row * DIM + lane * 8;
  float v[8]; float s = 0.0f;
  #pragma unroll
  for (int i = 0; i < 8; ++i) { v[i] = xr[i]; s += v[i]; }
  #pragma unroll
  for (int m = 16; m > 0; m >>= 1) s += __shfl_xor(s, m, 32);
  float mean = s * (1.0f / DIM);
  float vs = 0.0f;
  #pragma unroll
  for (int i = 0; i < 8; ++i) { float d = v[i] - mean; vs += d * d; }
  #pragma unroll
  for (int m = 16; m > 0; m >>= 1) vs += __shfl_xor(vs, m, 32);
  float inv = rsqrtf(vs * (1.0f / DIM) + 1e-5f);
  float* orow = o + (size_t)row * DIM + lane * 8;
  #pragma unroll
  for (int i = 0; i < 8; ++i)
    orow[i] = (v[i] - mean) * inv * g[lane * 8 + i] + b[lane * 8 + i];
}
// in-place row softmax, one 256-thread block per row
__global__ void softmax_rows_kernel(float* __restrict__ x, int cols) {
  __shared__ float red[256];
  int row = blockIdx.x, t = threadIdx.x;
  float* xr = x + (size_t)row * cols;
  float mx = -3.4e38f;
  for (int c = t; c < cols; c += 256) mx = fmaxf(mx, xr[c]);
  red[t] = mx; __syncthreads();
  for (int s = 128; s > 0; s >>= 1) { if (t < s) red[t] = fmaxf(red[t], red[t + s]); __syncthreads(); }
  mx = red[0]; __syncthreads();
  float sum = 0.0f;
  for (int c = t; c < cols; c += 256) { float e = expf(xr[c] - mx); xr[c] = e; sum += e; }
  red[t] = sum; __syncthreads();
  for (int s = 128; s > 0; s >>= 1) { if (t < s) red[t] += red[t + s]; __syncthreads(); }
  float inv = 1.0f / red[0];
  for (int c = t; c < cols; c += 256) xr[c] *= inv;
}
__global__ void softmax2_kernel(const float* __restrict__ logits, float* __restrict__ out) {
  int r = blockIdx.x * blockDim.x + threadIdx.x;
  if (r >= SP) return;
  float a = logits[r * 2], b = logits[r * 2 + 1];
  float m = fmaxf(a, b);
  float ea = expf(a - m), eb = expf(b - m);
  float inv = 1.0f / (ea + eb);
  out[r * 2] = ea * inv; out[r * 2 + 1] = eb * inv;
}

// ---------------- host orchestration ----------------
static inline void gemm(hipStream_t st, const float* A, int lda, long long sA,
                        const float* B, int ldb, long long sB, int tB,
                        float* C, int ldc, long long sC,
                        const float* bias, const float* addm, int ldadd, long long sAdd,
                        float alpha, int act, int M, int N, int K, int batch) {
  dim3 g((N + 127) / 128, (M + 63) / 64, batch);
  gemm_bf16_kernel<<<g, dim3(256), 0, st>>>(A, lda, sA, B, ldb, sB, tB, C, ldc, sC,
                                            bias, addm, ldadd, sAdd, alpha, act, M, N, K);
}

extern "C" void kernel_launch(void* const* d_in, const int* in_sizes, int n_in,
                              void* d_out, int out_size, void* d_ws, size_t ws_size,
                              hipStream_t stream) {
  (void)in_sizes; (void)n_in; (void)out_size; (void)ws_size;
  const float* T[2]     = {(const float*)d_in[0], (const float*)d_in[1]};
  const float* Q        = (const float*)d_in[2];
  const float* Aat      = (const float*)d_in[3];
  const float* convW_in = (const float*)d_in[4];
  const float* convB_in = (const float*)d_in[5];
  const float* convW    = (const float*)d_in[6];
  const float* convB    = (const float*)d_in[7];
  const float* fc0_W    = (const float*)d_in[8];
  const float* fc0_b    = (const float*)d_in[9];
  const float* fc_W     = (const float*)d_in[10];
  const float* fc_b     = (const float*)d_in[11];
  const float* pos      = (const float*)d_in[12];
  const float* ln1_g    = (const float*)d_in[13];
  const float* ln1_b    = (const float*)d_in[14];
  const float* qkv_W    = (const float*)d_in[15];
  const float* out_W    = (const float*)d_in[16];
  const float* out_b    = (const float*)d_in[17];
  const float* ln2_g    = (const float*)d_in[18];
  const float* ln2_b    = (const float*)d_in[19];
  const float* ff_W1    = (const float*)d_in[20];
  const float* ff_b1    = (const float*)d_in[21];
  const float* ff_W2    = (const float*)d_in[22];
  const float* ff_b2    = (const float*)d_in[23];
  const float* head_W1  = (const float*)d_in[24];
  const float* head_b1  = (const float*)d_in[25];
  const float* head_W2  = (const float*)d_in[26];
  const float* head_b2  = (const float*)d_in[27];
  float* out = (float*)d_out;

  // workspace carve (floats)
  float* w = (float*)d_ws;
  size_t off = 0;
  auto alloc = [&](size_t n) { float* p = w + off; off += n; return p; };
  float* cb0   = alloc((size_t)256 * HHW);
  float* cb1   = alloc((size_t)256 * HHW);
  float* cbt   = alloc((size_t)256 * HHW);
  float* LS[2] = {alloc((size_t)SP * BAND), alloc((size_t)SP * BAND)};
  int*   spidx = (int*)alloc(HHW);
  int*   counts= (int*)alloc(SP);
  float* pacc  = alloc((size_t)SP * 256);
  float* pout  = alloc((size_t)SP * 256);
  float* cat   = alloc((size_t)SP * 512);
  float* xc[2] = {alloc((size_t)SP * DIM), alloc((size_t)SP * DIM)};
  float* xo[2] = {alloc((size_t)SP * DIM), alloc((size_t)SP * DIM)};
  float* hbuf  = alloc((size_t)SP * DIM);
  float* qkvb  = alloc((size_t)SP * 3 * HEADS * DH);   // SP x 1536
  float* dots  = alloc((size_t)HEADS * SP * SP);
  float* obuf  = alloc((size_t)SP * HEADS * DH);       // SP x 512
  float* ffh   = alloc((size_t)SP * HID);
  float* dbuf  = alloc((size_t)SP * DIM);
  float* hh    = alloc((size_t)SP * 128);
  float* logit = alloc((size_t)SP * 2);

  float* conv_cur[2] = {cb0, cb1};
  float* conv_tmp = cbt;

  auto zf = [&](float* p, int n) { zero_f_kernel<<<(n + 255) / 256, 256, 0, stream>>>(p, n); };
  auto pool = [&](const float* X, int C, float* o) {
    zf(pacc, SP * C);
    pool_accum_kernel<<<(C * HHW + 255) / 256, 256, 0, stream>>>(X, spidx, pacc, C);
    ct_div_kernel<<<(SP * C + 255) / 256, 256, 0, stream>>>(pacc, counts, o, C);
  };
  auto swapx = [&](int br) { float* t = xc[br]; xc[br] = xo[br]; xo[br] = t; };

  auto encoder = [&](int s, int br) {
    size_t idx = (size_t)(s * 2 + br);
    addpos_kernel<<<(SP * DIM + 255) / 256, 256, 0, stream>>>(xc[br], pos + idx * SP * DIM, SP * DIM);
    ln_kernel<<<SP / 8, 256, 0, stream>>>(xc[br], ln1_g + idx * DIM, ln1_b + idx * DIM, hbuf);
    // qkv = h @ W (1024 x 1536)
    gemm(stream, hbuf, DIM, 0, qkv_W + idx * DIM * 1536, 1536, 0, 0,
         qkvb, 1536, 0, nullptr, nullptr, 0, 0, 1.0f, 0, SP, 1536, DIM, 1);
    // dots = scale * q k^T + A  (batched over heads, B transposed)
    gemm(stream, qkvb, 1536, 64, qkvb + HEADS * DH, 1536, 64, 1,
         dots, SP, (long long)SP * SP, nullptr, Aat, SP, 0,
         ATT_SCALE, 0, SP, SP, DH, HEADS);
    softmax_rows_kernel<<<HEADS * SP, 256, 0, stream>>>(dots, SP);
    // o = attn @ v
    gemm(stream, dots, SP, (long long)SP * SP, qkvb + 2 * HEADS * DH, 1536, 64, 0,
         obuf, HEADS * DH, 64, nullptr, nullptr, 0, 0, 1.0f, 0, SP, DH, SP, HEADS);
    // x = x + o @ oW + ob
    gemm(stream, obuf, HEADS * DH, 0, out_W + idx * HEADS * DH * DIM, DIM, 0, 0,
         xo[br], DIM, 0, out_b + idx * DIM, xc[br], DIM, 0, 1.0f, 0, SP, DIM, HEADS * DH, 1);
    swapx(br);
    ln_kernel<<<SP / 8, 256, 0, stream>>>(xc[br], ln2_g + idx * DIM, ln2_b + idx * DIM, hbuf);
    // ffh = gelu(h2 @ W1 + b1)
    gemm(stream, hbuf, DIM, 0, ff_W1 + idx * DIM * HID, HID, 0, 0,
         ffh, HID, 0, ff_b1 + idx * HID, nullptr, 0, 0, 1.0f, 2, SP, HID, DIM, 1);
    // x = x + ffh @ W2 + b2
    gemm(stream, ffh, HID, 0, ff_W2 + idx * HID * DIM, DIM, 0, 0,
         xo[br], DIM, 0, ff_b2 + idx * DIM, xc[br], DIM, 0, 1.0f, 0, SP, DIM, HID, 1);
    swapx(br);
  };

  // ---- superpixel assignment + counts ----
  zero_i_kernel<<<(SP + 255) / 256, 256, 0, stream>>>(counts, SP);
  q_index_kernel<<<(HHW + 255) / 256, 256, 0, stream>>>(Q, spidx, counts);

  // ---- LS[br] = ct(Q, T[br]) ----
  for (int br = 0; br < 2; ++br) pool(T[br], BAND, LS[br]);

  // ---- stage 0 ----
  for (int br = 0; br < 2; ++br) {
    conv3x3_bf16_kernel<<<dim3(256 / 128, HHW / 64), 256, 0, stream>>>(
        T[br], convW_in + (size_t)br * 256 * BAND * 9, convB_in + (size_t)br * 256,
        conv_cur[br], BAND, 256);
    pool(conv_cur[br], 256, pout);
    build_cat_kernel<<<(SP * (256 + BAND) + 255) / 256, 256, 0, stream>>>(cat, pout, 256, LS[br], BAND);
    gemm(stream, cat, 256 + BAND, 0, fc0_W + (size_t)br * (256 + BAND) * DIM, DIM, 0, 0,
         xc[br], DIM, 0, fc0_b + (size_t)br * DIM, nullptr, 0, 0,
         1.0f, 0, SP, DIM, 256 + BAND, 1);
    encoder(0, br);
  }

  // ---- stages 1..5 ----
  for (int s = 1; s < NS; ++s) {
    for (int br = 0; br < 2; ++br) {
      size_t ci = (size_t)((s - 1) * 2 + br);
      conv3x3_bf16_kernel<<<dim3(256 / 128, HHW / 64), 256, 0, stream>>>(
          conv_cur[br], convW + ci * 256 * 256 * 9, convB + ci * 256, conv_tmp, 256, 256);
      { float* t = conv_cur[br]; conv_cur[br] = conv_tmp; conv_tmp = t; }
      pool(conv_cur[br], 256, pout);
      build_cat_kernel<<<(SP * 512 + 255) / 256, 256, 0, stream>>>(cat, pout, 256, xc[br], DIM);
      gemm(stream, cat, 512, 0, fc_W + ci * 512 * DIM, DIM, 0, 0,
           xo[br], DIM, 0, fc_b + ci * DIM, nullptr, 0, 0, 1.0f, 0, SP, DIM, 512, 1);
      swapx(br);
      encoder(s, br);
    }
  }

  // ---- head ----
  absdiff_kernel<<<(SP * DIM + 255) / 256, 256, 0, stream>>>(xc[0], xc[1], dbuf, SP * DIM);
  gemm(stream, dbuf, DIM, 0, head_W1, 128, 0, 0, hh, 128, 0,
       head_b1, nullptr, 0, 0, 1.0f, 1, SP, 128, DIM, 1);
  gemm(stream, hh, 128, 0, head_W2, 2, 0, 0, logit, 2, 0,
       head_b2, nullptr, 0, 0, 1.0f, 0, SP, 2, 128, 1);
  softmax2_kernel<<<(SP + 255) / 256, 256, 0, stream>>>(logit, out);
}
